// MoEBlock_57758720196694
// MI455X (gfx1250) — matrix operand order repaired and checked by reading, not compiled, not run on hardware
//
#include <hip/hip_runtime.h>
#include <hip/hip_bf16.h>

typedef unsigned short u16;
typedef unsigned int   u32;

typedef __attribute__((ext_vector_type(16))) __bf16 v16bf;
typedef __attribute__((ext_vector_type(2)))  __bf16 bf2;
typedef __attribute__((ext_vector_type(8)))  float  v8f;
typedef __attribute__((ext_vector_type(4)))  unsigned int v4u;
typedef __attribute__((ext_vector_type(4)))  int v4i;
typedef __attribute__((ext_vector_type(8)))  int v8i;

#define EXPERTS 32
#define HID     2048
#define INTER   1024
#define GTOK    1024   // tokens per expert (32768 / 32)

#define BM   128       // token rows per block
#define BK   32        // K per WMMA step (bf16)
#define LSTR 40        // LDS row stride in halves (80B: 16B-aligned, staggered)

#if defined(__has_builtin)
#  if __has_builtin(__builtin_amdgcn_tensor_load_to_lds) && \
      __has_builtin(__builtin_amdgcn_tensor_store_from_lds) && \
      __has_builtin(__builtin_amdgcn_s_wait_tensorcnt)
#    define HAVE_TDM 1
#  endif
#endif
#ifndef HAVE_TDM
#  define HAVE_TDM 0
#endif

// ---------- helpers ----------

__device__ __forceinline__ u16 f2bf(float f) {   // native fp32 -> bf16 (RNE)
  __bf16 t = (__bf16)f;
  return __builtin_bit_cast(u16, t);
}
__device__ __forceinline__ u32 pack2(float lo, float hi) {
  bf2 t;
  t[0] = (__bf16)lo;
  t[1] = (__bf16)hi;   // pair -> packed cvt when available
  return __builtin_bit_cast(u32, t);
}

union FragBF { uint4 q[2]; v16bf v; };

// A-matrix 16x32 bf16 fragment from LDS tile (row-major [16][LSTR]).
// Contiguous K order used here (same as B): lane<16 -> row=lane, K 0..15; lane>=16 -> K 16..31.
__device__ __forceinline__ v16bf ldsA16x32(const u16* p0, int lane) {
  const u16* p = p0 + (lane & 15) * LSTR + ((lane >> 4) << 4);
  FragBF f;
  f.q[0] = *(const uint4*)(p);        // K = k0 .. k0+7
  f.q[1] = *(const uint4*)(p + 8);    // K = k0+8 .. k0+15
  return f.v;
}
// B-matrix 32x16 bf16 fragment from LDS stored transposed ([N][LSTR] over K).
// Contiguous K order used here (same as A): lane<16 -> col=lane, K 0..15; lane>=16 -> col=lane-16, K 16..31.
__device__ __forceinline__ v16bf ldsB32x16(const u16* p0, int lane) {
  const u16* p = p0 + (lane & 15) * LSTR + ((lane >> 4) << 4);
  FragBF f;
  f.q[0] = *(const uint4*)(p);        // K = k0 .. k0+7
  f.q[1] = *(const uint4*)(p + 8);    // K = k0+8 .. k0+15
  return f.v;
}

// ---------- kernel 1: gate_up = x @ w13, SwiGLU, h (bf16) ----------

__global__ __launch_bounds__(256)
void moe_gateup_kernel(const float* __restrict__ x,
                       const float* __restrict__ w13,
                       u16* __restrict__ h) {
  __shared__ __align__(16) u16 As[2][BM * LSTR];     // x tile  [128][32]
  __shared__ __align__(16) u16 Bt[2][128 * LSTR];    // w13^T   [64 gate + 64 up][32]
  __shared__ __align__(16) u16 Hs[BM * 64];          // dense h tile for bulk store

  const int e    = blockIdx.z;
  const int m0   = blockIdx.y * BM;
  const int n0   = blockIdx.x * 64;       // intermediate column base
  const int tid  = threadIdx.x;
  const int lane = tid & 31;
  const int wave = tid >> 5;
  const int wm   = wave >> 1;             // 0..3  (M)
  const int wn   = wave & 1;              // 0..1  (N)

  const float* xg   = x   + ((size_t)e * GTOK + m0) * HID;
  const float* w13e = w13 + (size_t)e * HID * (2 * INTER);

  v8f zero = {};
  v8f accG[2][2], accU[2][2];
#pragma unroll
  for (int i = 0; i < 2; ++i)
#pragma unroll
    for (int j = 0; j < 2; ++j) { accG[i][j] = zero; accU[i][j] = zero; }

  auto stage = [&](int kk, int buf) {
    const int kb = kk * BK;
    __builtin_prefetch(xg + (size_t)(tid >> 3) * HID + kb + BK, 0, 3);
    __builtin_prefetch(w13e + (size_t)(kb + BK) * (2 * INTER) + n0, 0, 3);
    // A: 128x32 fp32 -> bf16 LDS (1024 float4, 4 per thread)
#pragma unroll
    for (int i = 0; i < 4; ++i) {
      int f   = tid + i * 256;
      int row = f >> 3;                   // 8 float4 per row
      int c4  = (f & 7) << 2;
      const float4 v = *(const float4*)(xg + (size_t)row * HID + kb + c4);
      uint2 p; p.x = pack2(v.x, v.y); p.y = pack2(v.z, v.w);
      *(uint2*)&As[buf][row * LSTR + c4] = p;
    }
    // B: k-pair x 4-col blocks; pack (k,k+1) per column -> b32 LDS stores
#pragma unroll
    for (int i = 0; i < 2; ++i) {
      int f    = tid + i * 256;           // 0..511
      int k    = (f >> 5) << 1;           // even k row 0..30
      int cc   = (f & 31) << 2;           // combined col 0..124
      int gcol = (cc < 64) ? (n0 + cc) : (INTER + n0 + (cc - 64));
      const float4 r0 = *(const float4*)(w13e + (size_t)(kb + k)     * (2 * INTER) + gcol);
      const float4 r1 = *(const float4*)(w13e + (size_t)(kb + k + 1) * (2 * INTER) + gcol);
      *(u32*)&Bt[buf][(cc + 0) * LSTR + k] = pack2(r0.x, r1.x);
      *(u32*)&Bt[buf][(cc + 1) * LSTR + k] = pack2(r0.y, r1.y);
      *(u32*)&Bt[buf][(cc + 2) * LSTR + k] = pack2(r0.z, r1.z);
      *(u32*)&Bt[buf][(cc + 3) * LSTR + k] = pack2(r0.w, r1.w);
    }
  };

  auto mma_step = [&](int buf) {
    v16bf a[2], bg[2], bu[2];
#pragma unroll
    for (int fm = 0; fm < 2; ++fm)
      a[fm] = ldsA16x32(&As[buf][(wm * 32 + fm * 16) * LSTR], lane);
#pragma unroll
    for (int fn = 0; fn < 2; ++fn) {
      bg[fn] = ldsB32x16(&Bt[buf][(wn * 32 + fn * 16) * LSTR], lane);
      bu[fn] = ldsB32x16(&Bt[buf][(64 + wn * 32 + fn * 16) * LSTR], lane);
    }
#pragma unroll
    for (int fm = 0; fm < 2; ++fm)
#pragma unroll
      for (int fn = 0; fn < 2; ++fn) {
        accG[fm][fn] = __builtin_amdgcn_wmma_f32_16x16x32_bf16(
            false, a[fm], false, bg[fn], (short)0, accG[fm][fn], false, false);
        accU[fm][fn] = __builtin_amdgcn_wmma_f32_16x16x32_bf16(
            false, a[fm], false, bu[fn], (short)0, accU[fm][fn], false, false);
      }
  };

  const int KSTEPS = HID / BK;            // 64
  stage(0, 0);
  __syncthreads();
  for (int kk = 0; kk < KSTEPS; ++kk) {
    int buf = kk & 1;
    if (kk + 1 < KSTEPS) stage(kk + 1, buf ^ 1);
    mma_step(buf);
    __syncthreads();
  }

  // SwiGLU epilogue: h = silu(gate) * up -> dense LDS tile [128][64]
#pragma unroll
  for (int fm = 0; fm < 2; ++fm)
#pragma unroll
    for (int fn = 0; fn < 2; ++fn) {
      int rbase = wm * 32 + fm * 16 + ((lane >> 4) << 3);
      int lcol  = wn * 32 + fn * 16 + (lane & 15);
#pragma unroll
      for (int v = 0; v < 8; ++v) {
        float g = accG[fm][fn][v];
        float u = accU[fm][fn][v];
        float s = g / (1.0f + __expf(-g));
        Hs[(rbase + v) * 64 + lcol] = f2bf(s * u);
      }
    }
  __syncthreads();

  u16* hrow = h + ((size_t)e * GTOK + m0) * INTER;
#if HAVE_TDM
  if (wave == 0) {                         // bulk DMA: dense LDS tile -> h
    unsigned long long ga = (unsigned long long)(size_t)(hrow + n0);
    v4u g0;
    g0[0] = 1u;                                          // count=1, user mode
    g0[1] = (u32)(size_t)&Hs[0];                         // lds_addr
    g0[2] = (u32)ga;                                     // global_addr lo
    g0[3] = (u32)((ga >> 32) & 0x1FFFFFFu) | (2u << 30); // addr hi | type=2
    v8i g1;
    g1[0] = (int)(1u << 16);             // data_size = 2B, no padding
    g1[1] = (int)(1024u << 16);          // tensor_dim0 = INTER
    g1[2] = (int)0xFFFF0000u;            // tensor_dim1 = 65535 (no OOB clamp)
    g1[3] = (int)(64u << 16);            // tile_dim0 = 64 elements
    g1[4] = 128;                         // tile_dim1 = 128 rows
    g1[5] = 1024;                        // tensor_dim0_stride = INTER
    g1[6] = 0;
    g1[7] = 0;
    v4i gz = {0, 0, 0, 0};
#if __clang_major__ >= 23
    v8i gz8 = {0, 0, 0, 0, 0, 0, 0, 0};
    __builtin_amdgcn_tensor_store_from_lds(g0, g1, gz, gz, gz8, 0);
#else
    __builtin_amdgcn_tensor_store_from_lds(g0, g1, gz, gz, 0);
#endif
    __builtin_amdgcn_s_wait_tensorcnt(0);
  }
#else
  // fallback: coalesced 16B copies LDS -> h (1024 uint4, 4 per thread)
#pragma unroll
  for (int i = 0; i < 4; ++i) {
    int f   = tid + i * 256;
    int row = f >> 3;                    // 8 uint4 per row
    int c   = (f & 7) << 3;              // half offset
    *(uint4*)(hrow + (size_t)row * INTER + n0 + c) = *(const uint4*)&Hs[row * 64 + c];
  }
#endif
}

// ---------- kernel 2: out = h @ w2 ----------

__global__ __launch_bounds__(256)
void moe_down_kernel(const u16* __restrict__ h,
                     const float* __restrict__ w2,
                     float* __restrict__ out) {
  __shared__ __align__(16) u16 As[2][BM * LSTR];     // h tile  [128][32] (bf16)
  __shared__ __align__(16) u16 Bt[2][128 * LSTR];    // w2^T    [128][32]

  const int e    = blockIdx.z;
  const int m0   = blockIdx.y * BM;
  const int n0   = blockIdx.x * 128;      // hidden column base
  const int tid  = threadIdx.x;
  const int lane = tid & 31;
  const int wave = tid >> 5;
  const int wm   = wave >> 1;             // 0..3
  const int wn   = wave & 1;              // 0..1 (64 cols each)

  const u16*   hg  = h  + ((size_t)e * GTOK + m0) * INTER;
  const float* w2e = w2 + (size_t)e * INTER * HID;

  v8f zero = {};
  v8f acc[2][4];
#pragma unroll
  for (int i = 0; i < 2; ++i)
#pragma unroll
    for (int j = 0; j < 4; ++j) acc[i][j] = zero;

#if HAVE_TDM
  // TDM descriptor constants: 2D tile 32 halves x 128 rows out of the bf16 h
  // tensor (row stride INTER halves), data_size=2B, LDS padding 16B per 64B
  // of data -> LDS row stride = 40 halves = LSTR.
  const unsigned long long gaBase = (unsigned long long)(size_t)hg;
  const u32 ldsA0 = (u32)(size_t)&As[0][0];
  const u32 ldsA1 = (u32)(size_t)&As[1][0];
  v8i g1c;
  g1c[0] = (int)((1u << 16) |          // data_size = 1 -> 2 bytes
                 (1u << 20) |          // pad_enable
                 (3u << 22) |          // pad_interval: 16 DWORDs (64B)
                 (3u << 25));          // pad_amount:   4 DWORDs (16B)
  g1c[1] = (int)(1024u << 16);         // tensor_dim0 = INTER (bits 79:48 lo)
  g1c[2] = (int)0xFFFF0000u;           // tensor_dim1 = 65535 (no OOB clamp)
  g1c[3] = (int)(32u << 16);           // tile_dim0 = 32 elements (bits 127:112)
  g1c[4] = 128;                        // tile_dim1 = 128 rows
  g1c[5] = 1024;                       // tensor_dim0_stride = INTER
  g1c[6] = 0;
  g1c[7] = 0;
#endif

  auto stage = [&](int kk, int buf) {
    const int kb = kk * BK;
    __builtin_prefetch(w2e + (size_t)(kb + BK) * HID + n0, 0, 3);
#if HAVE_TDM
    if (wave == 0) {                     // one TDM op moves the whole A tile
      unsigned long long ga = gaBase + (unsigned long long)kb * 2ull;
      v4u g0;
      g0[0] = 1u;                                        // count=1, user mode
      g0[1] = buf ? ldsA1 : ldsA0;                       // lds_addr
      g0[2] = (u32)ga;                                   // global_addr lo
      g0[3] = (u32)((ga >> 32) & 0x1FFFFFFu) | (2u << 30); // addr hi | type=2
      v4i gz = {0, 0, 0, 0};
#if __clang_major__ >= 23
      v8i gz8 = {0, 0, 0, 0, 0, 0, 0, 0};
      __builtin_amdgcn_tensor_load_to_lds(g0, g1c, gz, gz, gz8, 0);
#else
      __builtin_amdgcn_tensor_load_to_lds(g0, g1c, gz, gz, 0);
#endif
    }
#else
    // A: 128x32 bf16, already bf16: straight 16B copies (512 uint4, 2/thread)
#pragma unroll
    for (int i = 0; i < 2; ++i) {
      int f   = tid + i * 256;
      int row = f >> 2;                   // 4 uint4 per row
      int c   = (f & 3) << 3;             // half offset
      uint4 v = *(const uint4*)(hg + (size_t)row * INTER + kb + c);
      *(uint4*)&As[buf][row * LSTR + c] = v;
    }
#endif
    // B: k-pair x 4-col blocks; pack (k,k+1) per column -> b32 LDS stores
#pragma unroll
    for (int i = 0; i < 2; ++i) {
      int f  = tid + i * 256;             // 0..511
      int k  = (f >> 5) << 1;             // even k row 0..30
      int cc = (f & 31) << 2;             // col 0..124
      const float4 r0 = *(const float4*)(w2e + (size_t)(kb + k)     * HID + n0 + cc);
      const float4 r1 = *(const float4*)(w2e + (size_t)(kb + k + 1) * HID + n0 + cc);
      *(u32*)&Bt[buf][(cc + 0) * LSTR + k] = pack2(r0.x, r1.x);
      *(u32*)&Bt[buf][(cc + 1) * LSTR + k] = pack2(r0.y, r1.y);
      *(u32*)&Bt[buf][(cc + 2) * LSTR + k] = pack2(r0.z, r1.z);
      *(u32*)&Bt[buf][(cc + 3) * LSTR + k] = pack2(r0.w, r1.w);
    }
  };

  auto mma_step = [&](int buf) {
    v16bf a[2], b[4];
#pragma unroll
    for (int fm = 0; fm < 2; ++fm)
      a[fm] = ldsA16x32(&As[buf][(wm * 32 + fm * 16) * LSTR], lane);
#pragma unroll
    for (int fn = 0; fn < 4; ++fn)
      b[fn] = ldsB32x16(&Bt[buf][(wn * 64 + fn * 16) * LSTR], lane);
#pragma unroll
    for (int fm = 0; fm < 2; ++fm)
#pragma unroll
      for (int fn = 0; fn < 4; ++fn)
        acc[fm][fn] = __builtin_amdgcn_wmma_f32_16x16x32_bf16(
            false, a[fm], false, b[fn], (short)0, acc[fm][fn], false, false);
  };

  const int KSTEPS = INTER / BK;          // 32
  stage(0, 0);
#if HAVE_TDM
  if (wave == 0) __builtin_amdgcn_s_wait_tensorcnt(0);
#endif
  __syncthreads();
  for (int kk = 0; kk < KSTEPS; ++kk) {
    int buf = kk & 1;
    if (kk + 1 < KSTEPS) stage(kk + 1, buf ^ 1);
    mma_step(buf);
#if HAVE_TDM
    if (wave == 0) __builtin_amdgcn_s_wait_tensorcnt(0);
#endif
    __syncthreads();
  }

  float* og = out + ((size_t)e * GTOK + m0) * HID;
#pragma unroll
  for (int fm = 0; fm < 2; ++fm)
#pragma unroll
    for (int fn = 0; fn < 4; ++fn) {
      int rbase = wm * 32 + fm * 16 + ((lane >> 4) << 3);
      int col   = n0 + wn * 64 + fn * 16 + (lane & 15);
#pragma unroll
      for (int v = 0; v < 8; ++v)
        og[(size_t)(rbase + v) * HID + col] = acc[fm][fn][v];
    }
}

// ---------- launcher ----------

extern "C" void kernel_launch(void* const* d_in, const int* in_sizes, int n_in,
                              void* d_out, int out_size, void* d_ws, size_t ws_size,
                              hipStream_t stream) {
  (void)in_sizes; (void)n_in; (void)out_size; (void)ws_size;
  const float* x   = (const float*)d_in[0];
  const float* w13 = (const float*)d_in[1];
  const float* w2  = (const float*)d_in[2];
  float* out = (float*)d_out;
  u16* h = (u16*)d_ws;                    // 32768 * 1024 bf16 = 64 MB scratch

  dim3 blk(256);
  dim3 g1(INTER / 64, GTOK / BM, EXPERTS);   // 16 x 8 x 32
  dim3 g2(HID / 128, GTOK / BM, EXPERTS);    // 16 x 8 x 32
  hipLaunchKernelGGL(moe_gateup_kernel, g1, blk, 0, stream, x, w13, h);
  hipLaunchKernelGGL(moe_down_kernel,   g2, blk, 0, stream, h, w2, out);
}
